// Ergodicity_Loss_12893491822607
// MI455X (gfx1250) — matrix operand order repaired and checked by reading, not compiled, not run on hardware
//
#include <hip/hip_runtime.h>
#include <hip/hip_bf16.h>

typedef __attribute__((ext_vector_type(16))) _Float16 v16h;
typedef __attribute__((ext_vector_type(8)))  _Float16 v8h;
typedef __attribute__((ext_vector_type(2)))  _Float16 h2;
typedef __attribute__((ext_vector_type(8)))  float    v8f;

#define KMAX 16
#define NAG  64
#define TT   512
#define BB   32
#define CH   8                 // K-chunks per batch -> 256 workgroups total
#define WAVES 8
#define ROUNDS 8               // per wave: 8 rounds x 64 samples = 512 = 4096/8 waves

// Wave-local LDS ordering: each wave only touches its own LDS slice, and the
// LDS pipeline executes a wave's DS ops in order (DScnt "in order"), so a
// compiler reordering fence is all that's needed -- no workgroup barrier.
#define WAVE_LDS_FENCE() asm volatile("" ::: "memory")

// Kernel 1: per-(b,chunk) partial C[16x16] via v_wmma_f32_16x16x32_f16,
// plus partial sum(u^2). Deterministic (no float atomics).
__global__ __launch_bounds__(256)
void ergo_main(const float* __restrict__ x, const float* __restrict__ u,
               float* __restrict__ wsC, float* __restrict__ wsU)
{
    // feature-major staging: [feature k][sample 0..63] halves, per wave
    __shared__ __align__(16) _Float16 cosA[WAVES][KMAX][64];
    __shared__ __align__(16) _Float16 cosB[WAVES][KMAX][64];
    __shared__ float red[WAVES * 256];

    const int tid  = threadIdx.x;
    const int lane = tid & 31;
    const int w    = tid >> 5;
    const int ch   = blockIdx.x;   // 0..7
    const int b    = blockIdx.y;   // 0..31

    const int m    = lane & 15;    // fragment row (A) / column (B)
    const int hsel = lane >> 4;

    // ---- kick off u^2 global loads early so they overlap the WMMA loop ----
    const int g = b * CH + ch;                 // workgroup id 0..255
    const float4* pu = (const float4*)u;
    const int base4 = g * 2048;                // 8192 floats per workgroup
    float4 uv[8];
#pragma unroll
    for (int r = 0; r < 8; ++r) uv[r] = pu[base4 + r * 256 + tid];

    v8f acc = {};

    for (int r = 0; r < ROUNDS; ++r) {
        // ---- stage 64 samples (one timestep row, lane owns n=2*lane,2*lane+1) ----
        const int t = ch * (TT / CH) + (w * ROUNDS + r);
        // float4 = {x0(s0), x1(s0), x0(s1), x1(s1)} : fully coalesced
        const float4 xv =
            ((const float4*)x)[(((size_t)t * BB + b) * NAG >> 1) + lane];

        // cos(pi*x) = v_cos(x/2)  (hardware cos takes revolutions)
        const float cA0 = __builtin_amdgcn_cosf(0.5f * xv.x);  // dim0, sample0
        const float cB0 = __builtin_amdgcn_cosf(0.5f * xv.y);  // dim1, sample0
        const float cA1 = __builtin_amdgcn_cosf(0.5f * xv.z);  // dim0, sample1
        const float cB1 = __builtin_amdgcn_cosf(0.5f * xv.w);  // dim1, sample1

        float a0p = 1.0f, a0c = cA0, a1p = 1.0f, a1c = cA1;
        float b0p = 1.0f, b0c = cB0, b1p = 1.0f, b1c = cB1;
        const float tA0 = 2.0f * cA0, tA1 = 2.0f * cA1;
        const float tB0 = 2.0f * cB0, tB1 = 2.0f * cB1;

        {   // k = 0, 1 : packed full-dword stores (two samples per dword)
            h2 p;
            p[0] = (_Float16)1.0f; p[1] = (_Float16)1.0f;
            *(h2*)&cosA[w][0][2 * lane] = p;
            *(h2*)&cosB[w][0][2 * lane] = p;
            p[0] = (_Float16)a0c;  p[1] = (_Float16)a1c;
            *(h2*)&cosA[w][1][2 * lane] = p;
            p[0] = (_Float16)b0c;  p[1] = (_Float16)b1c;
            *(h2*)&cosB[w][1][2 * lane] = p;
        }
#pragma unroll
        for (int k = 2; k < KMAX; ++k) {   // Chebyshev recurrence, 2 samples x 2 dims
            const float a0n = __builtin_fmaf(tA0, a0c, -a0p);
            const float a1n = __builtin_fmaf(tA1, a1c, -a1p);
            const float b0n = __builtin_fmaf(tB0, b0c, -b0p);
            const float b1n = __builtin_fmaf(tB1, b1c, -b1p);
            h2 pa; pa[0] = (_Float16)a0n; pa[1] = (_Float16)a1n;
            h2 pb; pb[0] = (_Float16)b0n; pb[1] = (_Float16)b1n;
            *(h2*)&cosA[w][k][2 * lane] = pa;
            *(h2*)&cosB[w][k][2 * lane] = pb;
            a0p = a0c; a0c = a0n;  a1p = a1c; a1c = a1n;
            b0p = b0c; b0c = b0n;  b1p = b1c; b1c = b1n;
        }
        WAVE_LDS_FENCE();   // keep stores before loads; HW LDS is in-order per wave

        // ---- two K=32 WMMA sub-blocks from the 64 staged samples ----
#pragma unroll
        for (int jb = 0; jb < 2; ++jb) {
            const int ko = jb * 32;
            // A: lanes 0-15 M=lane K={0..7,16..23}; lanes 16-31 K={8..15,24..31}
            const v8h alo = *(const v8h*)&cosA[w][m][ko + hsel * 8];
            const v8h ahi = *(const v8h*)&cosA[w][m][ko + 16 + hsel * 8];
            // B: lanes 0-15 N=lane K={0..7,16..23}; lanes 16-31 N=lane-16 K={8..15,24..31}
            const v8h blo = *(const v8h*)&cosB[w][m][ko + hsel * 8];
            const v8h bhi = *(const v8h*)&cosB[w][m][ko + 16 + hsel * 8];
            v16h afrag, bfrag;
#pragma unroll
            for (int h = 0; h < 8; ++h) {
                afrag[h] = alo[h]; afrag[8 + h] = ahi[h];
                bfrag[h] = blo[h]; bfrag[8 + h] = bhi[h];
            }
            acc = __builtin_amdgcn_wmma_f32_16x16x32_f16(
                      false, afrag, false, bfrag, (short)0, acc, false, false);
        }
        WAVE_LDS_FENCE();   // loads ordered before next round's stores
    }

    // ---- combine 8 waves' accumulators; element index = e*32 + lane ----
#pragma unroll
    for (int e = 0; e < 8; ++e) red[w * 256 + e * 32 + lane] = acc[e];
    __syncthreads();
    float s = 0.0f;
#pragma unroll
    for (int ww = 0; ww < WAVES; ++ww) s += red[ww * 256 + tid];
    wsC[g * 256 + tid] = s;

    // ---- partial sum(u^2): consume the early loads ----
    float us = 0.0f;
#pragma unroll
    for (int r = 0; r < 8; ++r) {
        const float4 v = uv[r];
        us = fmaf(v.x, v.x, us); us = fmaf(v.y, v.y, us);
        us = fmaf(v.z, v.z, us); us = fmaf(v.w, v.w, us);
    }
    __syncthreads();           // everyone done reading red[] above
    red[tid] = us;
    __syncthreads();
    for (int off = 128; off > 0; off >>= 1) {
        if (tid < off) red[tid] += red[tid + off];
        __syncthreads();
    }
    if (tid == 0) wsU[g] = red[0];
}

// Kernel 2: fold partials, apply analytic coeffs_density + normalization, emit loss.
__global__ __launch_bounds__(256)
void ergo_final(const float* __restrict__ wsC, const float* __restrict__ wsU,
                float* __restrict__ out)
{
    __shared__ float red[256];
    const int t = threadIdx.x;
    const int l = t & 31, e = t >> 5;
    const int i = e + 8 * (l >> 4);   // C-matrix row  (feature of dim 0)
    const int j = l & 15;             // C-matrix col  (feature of dim 1)

    const float PI = 3.14159265358979323846f;
    // v_k = 1 (k=0), 0 (k even>0), 2i/(pi k) (k odd) -> Re(v_i v_j):
    const float fi = (i & 1) ? 2.0f / (PI * (float)i) : 0.0f;
    const float fj = (j & 1) ? 2.0f / (PI * (float)j) : 0.0f;
    const float cdraw = (i == 0 && j == 0) ? 1.0f
                      : (((i & 1) && (j & 1)) ? -fi * fj : 0.0f);
    const float normj = (j == 0) ? 1.0f : 0.70710678118654752f;  // sqrt(L/2), only last dim
    const float cd = cdraw / normj;                               // coeffs_density
    const float invscale = 1.0f / (normj * (float)(NAG * TT));    // 1/(norm * N*T)

    float lsum = 0.0f;
    for (int b = 0; b < BB; ++b) {
        float s = 0.0f;
#pragma unroll
        for (int c = 0; c < CH; ++c) s += wsC[(b * CH + c) * 256 + t];
        const float diff = s * invscale - cd;
        lsum = fmaf(diff, diff, lsum);
    }
    red[t] = lsum;
    __syncthreads();
    for (int off = 128; off > 0; off >>= 1) {
        if (t < off) red[t] += red[t + off];
        __syncthreads();
    }
    if (t == 0) {
        float usum = 0.0f;
        for (int g = 0; g < BB * CH; ++g) usum += wsU[g];
        const float loss1 = red[0] / ((float)BB * 256.0f);
        const float loss2 = 1.0e-3f * usum /
                            (2.0f * (float)NAG * (float)TT * (float)BB);
        out[0] = loss1 + loss2;
    }
}

extern "C" void kernel_launch(void* const* d_in, const int* in_sizes, int n_in,
                              void* d_out, int out_size, void* d_ws, size_t ws_size,
                              hipStream_t stream) {
    const float* x = (const float*)d_in[0];   // [T,B,N,2] f32
    const float* u = (const float*)d_in[1];   // [T,B,N,2] f32
    float* wsC = (float*)d_ws;                // 256 workgroups * 256 floats
    float* wsU = wsC + 256 * 256;             // 256 floats
    float* out = (float*)d_out;

    dim3 grid(CH, BB);
    ergo_main<<<grid, 256, 0, stream>>>(x, u, wsC, wsU);
    ergo_final<<<1, 256, 0, stream>>>(wsC, wsU, out);
    (void)in_sizes; (void)n_in; (void)out_size; (void)ws_size;
}
